// DeltaNet_31877247271540
// MI455X (gfx1250) — compile-verified, matrix operand order checked by reading, not run on hardware
//
#include <hip/hip_runtime.h>

// ---------------------------------------------------------------- constants
#define Bn    2
#define Ln    2048
#define Dn    1024
#define Hn    4
#define DKn   256
#define DVn   256
#define CHn   32
#define NCHn  (Ln / CHn)        // 64 chunks
#define TOKn  (Bn * Ln)         // 4096 tokens

typedef __attribute__((ext_vector_type(16))) __bf16 v16bf;
typedef __attribute__((ext_vector_type(8)))  float  v8f;

// ---------------------------------------------------------------- WMMA utils
__device__ __forceinline__ v8f wmma_bf16(v16bf a, v16bf b, v8f c) {
  // D = A(16x32 bf16) x B(32x16 bf16) + C(16x16 f32); wave32
  return __builtin_amdgcn_wmma_f32_16x16x32_bf16(false, a, false, b, (short)0, c,
                                                 false, false);
}

// element e of a 16-bit A/B fragment -> K index (ISA 7.12.2, 16-bit 16x32 A layout)
__device__ __forceinline__ int frag_k(int e, int kb) {
  int v = e >> 1, h = e & 1;
  return (v < 4) ? (kb + 2 * v + h) : (16 + kb + 2 * (v - 4) + h);
}

// A fragment (M=16,K=32) from row-major bf16 tile: per-lane data is two
// contiguous 16B runs (k = kb..kb+7 and 16+kb..16+kb+7) -> 2 x b128 loads.
__device__ __forceinline__ v16bf fragA_vec(const __bf16* t, int ld) {
  int lane = threadIdx.x & 31;
  int m = lane & 15;
  int kb = (lane & 16) ? 8 : 0;
  const __bf16* p = t + (size_t)m * ld + kb;
  union { v16bf v; uint4 q[2]; } u;
  u.q[0] = *(const uint4*)(p);
  u.q[1] = *(const uint4*)(p + 16);
  return u.v;
}

// B fragment (K=32,N=16) from an N-major (transposed) f32 tile t[n*ldk + k]:
// 4 x float4 loads + pack-convert to bf16.
__device__ __forceinline__ v16bf fragB_f32T(const float* t, int ldk) {
  int lane = threadIdx.x & 31;
  int n = lane & 15;
  int kb = (lane & 16) ? 8 : 0;
  const float* p = t + (size_t)n * ldk + kb;
  float4 f0 = *(const float4*)(p);
  float4 f1 = *(const float4*)(p + 4);
  float4 f2 = *(const float4*)(p + 16);
  float4 f3 = *(const float4*)(p + 20);
  v16bf b;
  b[0]  = (__bf16)f0.x; b[1]  = (__bf16)f0.y; b[2]  = (__bf16)f0.z; b[3]  = (__bf16)f0.w;
  b[4]  = (__bf16)f1.x; b[5]  = (__bf16)f1.y; b[6]  = (__bf16)f1.z; b[7]  = (__bf16)f1.w;
  b[8]  = (__bf16)f2.x; b[9]  = (__bf16)f2.y; b[10] = (__bf16)f2.z; b[11] = (__bf16)f2.w;
  b[12] = (__bf16)f3.x; b[13] = (__bf16)f3.y; b[14] = (__bf16)f3.z; b[15] = (__bf16)f3.w;
  return b;
}

// B fragment (K=32,N=16) scalar gather from row-major bf16 tile (fallback)
__device__ __forceinline__ v16bf fragB_bf16(const __bf16* t, int ld) {
  int lane = threadIdx.x & 31;
  int n = lane & 15;
  int kb = (lane & 16) ? 8 : 0;
  v16bf b;
#pragma unroll
  for (int e = 0; e < 16; ++e) b[e] = t[(size_t)frag_k(e, kb) * ld + n];
  return b;
}

__device__ __forceinline__ unsigned lds_off(const void* p) {
  // generic address of LDS object: low 32 bits are the LDS byte offset
  return (unsigned)(size_t)p;
}

// async copy of one 16B packet: global -> LDS (ASYNCcnt-tracked)
__device__ __forceinline__ void async_b128(unsigned lds_addr, const void* gaddr) {
  asm volatile("global_load_async_to_lds_b128 %0, %1, off"
               :: "v"(lds_addr), "v"(gaddr) : "memory");
}
__device__ __forceinline__ void wait_async0() {
  asm volatile("s_wait_asynccnt 0x0" ::: "memory");
}

__device__ __forceinline__ float sigmoidf_(float x) { return 1.f / (1.f + expf(-x)); }
__device__ __forceinline__ float siluf_(float x)    { return x * sigmoidf_(x); }
__device__ __forceinline__ float geluf_(float x)    { return 0.5f * x * (1.f + erff(x * 0.70710678118f)); }

// ---------------------------------------------------------------- GEMM (bf16 WMMA)
// C[M,N] = act(A[M,K] * B[K,N] + bias). M % 64 == 0, K % 32 == 0; N guarded.
__launch_bounds__(256)
__global__ void gemm_bf16_kernel(const float* __restrict__ A, const float* __restrict__ Bm,
                                 const float* __restrict__ bias, float* __restrict__ C,
                                 int M, int N, int K, int act) {
  __shared__ __align__(16) __bf16 sA[64 * 32];    // row-major 64x32
  __shared__ __align__(16) __bf16 sBt[64 * 32];   // transposed: [n][k]
  const int tid  = threadIdx.x;
  const int lane = tid & 31;
  const int warp = tid >> 5;
  const int m0 = blockIdx.y * 64;
  const int n0 = blockIdx.x * 64;
  const int wm = warp >> 1;  // 0..3 : 16-row band
  const int wn = warp & 1;   // 0..1 : 32-col band
  v8f acc[2] = {{}, {}};

  for (int k0 = 0; k0 < K; k0 += 32) {
    // stage A 64x32 (float4 loads, f32 -> bf16)
#pragma unroll
    for (int i = 0; i < 2; ++i) {
      int idx = tid + i * 256;
      int r = idx >> 3, c4 = (idx & 7) << 2;
      const float* src = A + (size_t)(m0 + r) * K + k0 + c4;
      float4 f = *(const float4*)src;
      __bf16* dst = sA + r * 32 + c4;
      dst[0] = (__bf16)f.x; dst[1] = (__bf16)f.y;
      dst[2] = (__bf16)f.z; dst[3] = (__bf16)f.w;
      if (k0 + 32 < K) __builtin_prefetch(src + 32, 0, 0);  // global_prefetch_b8
    }
    // stage B 32x64 -> transposed LDS [n][k] (coalesced reads, N-guarded)
#pragma unroll
    for (int i = 0; i < 8; ++i) {
      int idx = tid + i * 256;
      int r = idx >> 6, c = idx & 63;
      float v = (n0 + c < N) ? Bm[(size_t)(k0 + r) * N + n0 + c] : 0.0f;
      sBt[c * 32 + r] = (__bf16)v;
    }
    __syncthreads();
    v16bf fa  = fragA_vec(sA + wm * 16 * 32, 32);
    v16bf fb0 = fragA_vec(sBt + (wn * 32) * 32, 32);       // [n][k] -> B frag
    v16bf fb1 = fragA_vec(sBt + (wn * 32 + 16) * 32, 32);
    acc[0] = wmma_bf16(fa, fb0, acc[0]);
    acc[1] = wmma_bf16(fa, fb1, acc[1]);
    __syncthreads();
  }

  const int nlo = lane & 15;
  const int mb  = (lane & 16) ? 8 : 0;
#pragma unroll
  for (int t = 0; t < 2; ++t) {
    int n = n0 + wn * 32 + t * 16 + nlo;
    if (n >= N) continue;
#pragma unroll
    for (int r = 0; r < 8; ++r) {
      int m = m0 + wm * 16 + mb + r;
      float v = acc[t][r] + (bias ? bias[n] : 0.f);
      if (act == 1) v = geluf_(v);
      C[(size_t)m * N + n] = v;
    }
  }
}

// ---------------------------------------------------------------- conv / elementwise
__global__ void dwconv4_silu_kernel(const float* __restrict__ x, const float* __restrict__ w,
                                    float* __restrict__ y) {
  int idx = blockIdx.x * blockDim.x + threadIdx.x;
  if (idx >= TOKn * Dn) return;
  int c = idx & (Dn - 1);
  int tok = idx >> 10;
  int l = tok & (Ln - 1);
  float acc = 0.f;
#pragma unroll
  for (int j = 0; j < 4; ++j) {
    int li = l + j - 3;
    if (li >= 0) acc += x[(size_t)(tok + j - 3) * Dn + c] * w[c * 4 + j];
  }
  y[idx] = siluf_(acc);
}

__global__ void fir_kernel(const float* __restrict__ x, const float* __restrict__ filt,
                           float* __restrict__ y, int K) {
  int idx = blockIdx.x * blockDim.x + threadIdx.x;
  if (idx >= TOKn * Dn) return;
  int c = idx & (Dn - 1);
  int tok = idx >> 10;
  int l = tok & (Ln - 1);
  float acc = 0.f;
  for (int j = 0; j < K; ++j) {
    int li = l + j - (K - 1);
    if (li >= 0) acc += x[(size_t)(tok + j - (K - 1)) * Dn + c] * filt[c * K + j];
  }
  y[idx] = acc;
}

__global__ void beta_kernel(const float* __restrict__ hidden, const float* __restrict__ Wb,
                            float* __restrict__ beta) {
  int wid  = (blockIdx.x * blockDim.x + threadIdx.x) >> 5;
  int lane = threadIdx.x & 31;
  if (wid >= TOKn * Hn) return;
  int h = wid & 3, tok = wid >> 2;
  const float* row = hidden + (size_t)tok * Dn;
  float s = 0.f;
  for (int d = lane; d < Dn; d += 32) s += row[d] * Wb[d * Hn + h];
#pragma unroll
  for (int o = 16; o; o >>= 1) s += __shfl_xor(s, o, 32);
  if (lane == 0) beta[tok * Hn + h] = sigmoidf_(s);
}

// l2norm q,k per (b,h,l); produce bf16 qn,kn,kb,vb in [B,H,L,256]
__global__ void qknorm_kernel(const float* __restrict__ q, const float* __restrict__ k,
                              const float* __restrict__ v, const float* __restrict__ beta,
                              __bf16* __restrict__ qn, __bf16* __restrict__ kn,
                              __bf16* __restrict__ kb, __bf16* __restrict__ vb) {
  int wid  = (blockIdx.x * blockDim.x + threadIdx.x) >> 5;
  int lane = threadIdx.x & 31;
  if (wid >= TOKn * Hn) return;
  int h = wid & 3, tok = wid >> 2;
  int b = tok >> 11, l = tok & (Ln - 1);
  const float* qr = q + (size_t)tok * Dn + h * DKn;
  const float* kr = k + (size_t)tok * Dn + h * DKn;
  const float* vr = v + (size_t)tok * Dn + h * DVn;
  float qv[8], kv[8], vv[8];
  float sq = 0.f, sk = 0.f;
#pragma unroll
  for (int j = 0; j < 8; ++j) {
    int d = lane + 32 * j;
    qv[j] = qr[d]; kv[j] = kr[d]; vv[j] = vr[d];
    sq += qv[j] * qv[j]; sk += kv[j] * kv[j];
  }
#pragma unroll
  for (int o = 16; o; o >>= 1) { sq += __shfl_xor(sq, o, 32); sk += __shfl_xor(sk, o, 32); }
  float qi = rsqrtf(sq + 1e-6f), ki = rsqrtf(sk + 1e-6f);
  float bet = beta[tok * Hn + h];
  size_t base = (((size_t)(b * Hn + h)) * Ln + l) * DKn;
#pragma unroll
  for (int j = 0; j < 8; ++j) {
    int d = lane + 32 * j;
    float knv = kv[j] * ki;
    qn[base + d] = (__bf16)(qv[j] * qi);
    kn[base + d] = (__bf16)knv;
    kb[base + d] = (__bf16)(knv * bet);
    vb[base + d] = (__bf16)(vv[j] * bet);
  }
}

// ---------------------------------------------------------------- chunk prepare
// Per (b,h,chunk): T = fwd-subst(-tril(kb kn^T,-1)) + I ; u = T vb ; w = T kb
__launch_bounds__(256)
__global__ void chunk_prepare_kernel(const __bf16* __restrict__ kn, const __bf16* __restrict__ kb,
                                     const __bf16* __restrict__ vb,
                                     __bf16* __restrict__ u_g, __bf16* __restrict__ w_g) {
  __shared__ __align__(16) __bf16 sKn[32 * 256];
  __shared__ __align__(16) __bf16 sKb[32 * 256];
  __shared__ __align__(16) __bf16 sVb[32 * 256];
  __shared__ float  sT[32 * 33];
  __shared__ __align__(16) __bf16 sTb[32 * 32];

  const int tid = threadIdx.x, lane = tid & 31, warp = tid >> 5;
  const int blk = blockIdx.x;           // bh*64 + chunk
  const int chunk = blk & 63;
  const int bh = blk >> 6;
  const size_t rowBase = ((size_t)bh * Ln + chunk * 32) * 256;

  // stage 3 x (32x256) bf16 tiles via async global->LDS b128 packets
  {
    const uint4* g0 = (const uint4*)(kn + rowBase);
    const uint4* g1 = (const uint4*)(kb + rowBase);
    const uint4* g2 = (const uint4*)(vb + rowBase);
    unsigned l0 = lds_off(sKn), l1 = lds_off(sKb), l2 = lds_off(sVb);
#pragma unroll
    for (int i = 0; i < 4; ++i) {
      int idx = tid + i * 256;                 // 16B packet index (1024 per tile)
      unsigned boff = (unsigned)idx * 16u;
      async_b128(l0 + boff, g0 + idx);
      async_b128(l1 + boff, g1 + idx);
      async_b128(l2 + boff, g2 + idx);
    }
    wait_async0();
  }
  __syncthreads();

  // M = kb kn^T (32x32, K=256): 4 tiles on waves 0..3. kn^T B-fragment == A-gather on kn.
  if (warp < 4) {
    int tm = warp >> 1, tn = warp & 1;
    v8f acc = {};
#pragma unroll
    for (int ks = 0; ks < 8; ++ks)
      acc = wmma_bf16(fragA_vec(sKb + tm * 16 * 256 + ks * 32, 256),
                      fragA_vec(sKn + tn * 16 * 256 + ks * 32, 256), acc);
    int n = lane & 15, mb = (lane & 16) ? 8 : 0;
#pragma unroll
    for (int r = 0; r < 8; ++r) {
      int i = tm * 16 + mb + r, j = tn * 16 + n;
      sT[i * 33 + j] = (j < i) ? -acc[r] : 0.f;
    }
  }
  __syncthreads();

  // forward substitution (reference fori_loop), one wave
  if (tid < 32) {
    int j = tid;
    for (int i = 1; i < 32; ++i) {
      float upd = 0.f;
      for (int c = 0; c < i; ++c) upd += sT[i * 33 + c] * sT[c * 33 + j];
      if (j < i) sT[i * 33 + j] += upd;
    }
    sT[j * 33 + j] = 1.0f;   // T += I
  }
  __syncthreads();
#pragma unroll
  for (int i = 0; i < 4; ++i) {
    int idx = tid + i * 256;
    sTb[idx] = (__bf16)sT[(idx >> 5) * 33 + (idx & 31)];
  }
  __syncthreads();

  // u = T vb, w = T kb : 64 tiles (16x16, K=32) over 8 waves
  int n = lane & 15, mb = (lane & 16) ? 8 : 0;
#pragma unroll
  for (int i = 0; i < 8; ++i) {
    int id = warp * 8 + i;
    int isW = id >> 5;
    int t = id & 31;
    int tm = t >> 4, tn = t & 15;
    const __bf16* src = isW ? sKb : sVb;
    v8f z = {};
    v8f acc = wmma_bf16(fragA_vec(sTb + tm * 16 * 32, 32),
                        fragB_bf16(src + tn * 16, 256), z);
    __bf16* dst = (isW ? w_g : u_g) + rowBase;
#pragma unroll
    for (int r = 0; r < 8; ++r)
      dst[(size_t)(tm * 16 + mb + r) * 256 + tn * 16 + n] = (__bf16)acc[r];
  }
}

// ---------------------------------------------------------------- delta scan
// Per (b,h,dv-slice of 32): S^T (32x256 f32, [v][dk]) lives in LDS across the scan.
__launch_bounds__(256)
__global__ void delta_scan_kernel(const __bf16* __restrict__ qn, const __bf16* __restrict__ kn,
                                  const __bf16* __restrict__ u_g, const __bf16* __restrict__ w_g,
                                  const float* __restrict__ forget_param,
                                  float* __restrict__ delta_out) {
  __shared__ float  sSt[32 * 256];      // S transposed: [v-slice n][dk m], 32 KB
  __shared__ float  sAttn[32 * 33];
  __shared__ __align__(16) __bf16 sAttnB[32 * 32];   // row-major [m][c]
  __shared__ __align__(16) __bf16 sUt[32 * 32];      // u_adj transposed: [n][c]
  __shared__ __align__(16) __bf16 sKT[256 * 32];     // k^T chunk: [dk][c], 16 KB

  const int tid = threadIdx.x, lane = tid & 31, warp = tid >> 5;
  const int blk = blockIdx.x;           // bh*8 + vq
  const int vq = blk & 7, bh = blk >> 3;
  const int b = bh >> 2, h = bh & 3;
  const int v0 = vq * 32;
  const float lam = 0.55f + 0.45f * sigmoidf_(forget_param[h]);
  const size_t bhL = (size_t)bh * Ln;

  for (int i = tid; i < 32 * 256; i += 256) sSt[i] = 0.f;
  __syncthreads();

  const int nlo = lane & 15, mb = (lane & 16) ? 8 : 0;

  for (int ch = 0; ch < NCHn; ++ch) {
    const int t0 = ch * 32;
    // stage k^T tile for this chunk (coalesced reads, transposed LDS writes)
    {
      const uint4* kr = (const uint4*)(kn + (bhL + t0) * 256);
#pragma unroll
      for (int i = 0; i < 4; ++i) {
        int idx = tid + i * 256;        // 1024 packets of 8 bf16
        int c = idx >> 5, d0 = (idx & 31) * 8;
        uint4 pkt = kr[idx];
        const __bf16* e = (const __bf16*)&pkt;
#pragma unroll
        for (int j = 0; j < 8; ++j) sKT[(d0 + j) * 32 + c] = e[j];
      }
    }
    __syncthreads();

    v8f accWS = {};
    // phase 1: waves 0..3 -> w·S ; waves 4..7 -> masked attn = tril(q k^T)
    if (warp < 4) {
      int tm = warp >> 1, tn = warp & 1;
      const __bf16* wb = w_g + (bhL + t0 + tm * 16) * 256;
#pragma unroll
      for (int ks = 0; ks < 8; ++ks)
        accWS = wmma_bf16(fragA_vec(wb + ks * 32, 256),
                          fragB_f32T(sSt + (tn * 16) * 256 + ks * 32, 256), accWS);
    } else {
      int w4 = warp - 4;
      int tm = w4 >> 1, tn = w4 & 1;
      v8f acc = {};
      const __bf16* qb = qn + (bhL + t0 + tm * 16) * 256;
      const __bf16* kb2 = kn + (bhL + t0 + tn * 16) * 256;
#pragma unroll
      for (int ks = 0; ks < 8; ++ks)
        acc = wmma_bf16(fragA_vec(qb + ks * 32, 256),
                        fragA_vec(kb2 + ks * 32, 256), acc);
#pragma unroll
      for (int r = 0; r < 8; ++r) {
        int i = tm * 16 + mb + r, j = tn * 16 + nlo;
        sAttn[i * 33 + j] = (j <= i) ? acc[r] : 0.f;
      }
    }
    __syncthreads();
    // phase 2: u_adj = u - w·S  -> sUt (transposed, bf16); bf16-ify attn
    if (warp < 4) {
      int tm = warp >> 1, tn = warp & 1;
      const __bf16* ub = u_g + (bhL + t0) * 256 + v0;
#pragma unroll
      for (int r = 0; r < 8; ++r) {
        int m = tm * 16 + mb + r, nn = tn * 16 + nlo;
        float val = (float)ub[(size_t)m * 256 + nn] - accWS[r];
        sUt[nn * 32 + m] = (__bf16)val;
      }
    } else {
      for (int i = tid - 128; i < 1024; i += 128)
        sAttnB[i] = (__bf16)sAttn[(i >> 5) * 33 + (i & 31)];
    }
    __syncthreads();
    // phase 3: o = q·S + attn·u_adj  -> delta_out [B,L,H*DV]
    if (warp < 4) {
      int tm = warp >> 1, tn = warp & 1;
      v8f acc = {};
      const __bf16* qb = qn + (bhL + t0 + tm * 16) * 256;
#pragma unroll
      for (int ks = 0; ks < 8; ++ks)
        acc = wmma_bf16(fragA_vec(qb + ks * 32, 256),
                        fragB_f32T(sSt + (tn * 16) * 256 + ks * 32, 256), acc);
      acc = wmma_bf16(fragA_vec(sAttnB + tm * 16 * 32, 32),
                      fragA_vec(sUt + (tn * 16) * 32, 32), acc);
#pragma unroll
      for (int r = 0; r < 8; ++r) {
        int m = tm * 16 + mb + r, nn = tn * 16 + nlo;
        size_t row = (size_t)b * Ln + t0 + m;
        delta_out[row * (Hn * DVn) + h * DVn + v0 + nn] = acc[r];
      }
    }
    __syncthreads();
    // phase 4: S = lam*S + k^T·u_adj ; C-operand carries lam*S; vector RMW of S^T
#pragma unroll
    for (int i = 0; i < 4; ++i) {
      int id = warp * 4 + i;            // 0..31
      int tm = id >> 1, tn = id & 1;    // tm: dk/16, tn: dv-slice/16
      v16bf a  = fragA_vec(sKT + (tm * 16) * 32, 32);        // k^T [m][c]
      v16bf bf = fragA_vec(sUt + (tn * 16) * 32, 32);        // u_adj^T [n][c]
      float* cp = sSt + (size_t)(tn * 16 + nlo) * 256 + tm * 16 + mb;
      float4 c0 = *(const float4*)cp;
      float4 c1 = *(const float4*)(cp + 4);
      v8f c;
      c[0] = lam * c0.x; c[1] = lam * c0.y; c[2] = lam * c0.z; c[3] = lam * c0.w;
      c[4] = lam * c1.x; c[5] = lam * c1.y; c[6] = lam * c1.z; c[7] = lam * c1.w;
      v8f outp = wmma_bf16(a, bf, c);
      float4 o0 = {outp[0], outp[1], outp[2], outp[3]};
      float4 o1 = {outp[4], outp[5], outp[6], outp[7]};
      *(float4*)cp = o0;
      *(float4*)(cp + 4) = o1;
    }
    __syncthreads();
  }
}

// ---------------------------------------------------------------- gate pipeline
__launch_bounds__(256)
__global__ void gate_build_kernel(const float* __restrict__ hidden, const float* __restrict__ sh,
                                  const float* __restrict__ lo, const float* __restrict__ de,
                                  const float* __restrict__ vd, float* __restrict__ gate_in) {
  int tok = blockIdx.x;
  int tid = threadIdx.x, lane = tid & 31, warp = tid >> 5;
  const float4* hs = (const float4*)(hidden + (size_t)tok * Dn);
  float4* gi4 = (float4*)(gate_in + (size_t)tok * 1088);
  gi4[tid] = hs[tid];                                  // 1024 f32
  const float* srcs[4] = {sh, lo, de, vd};
#pragma unroll
  for (int i = 0; i < 2; ++i) {
    int task = warp * 2 + i;                           // 0..15
    int tIdx = task >> 2, h = task & 3;
    const float* p = srcs[tIdx] + (size_t)tok * Dn + h * DVn;
    float s = 0.f, ss = 0.f, mx = -3.0e38f;
#pragma unroll
    for (int j = 0; j < 8; ++j) {
      float x = p[lane + 32 * j];
      s += x; ss += x * x; mx = fmaxf(mx, x);
    }
#pragma unroll
    for (int o = 16; o; o >>= 1) {
      s += __shfl_xor(s, o, 32);
      ss += __shfl_xor(ss, o, 32);
      mx = fmaxf(mx, __shfl_xor(mx, o, 32));
    }
    if (lane == 0) {
      float m = s * (1.f / 256.f);
      float var = ss * (1.f / 256.f) - m * m;
      float* o = gate_in + (size_t)tok * 1088 + 1024 + tIdx * 16 + h * 4;
      o[0] = m; o[1] = var; o[2] = mx; o[3] = sqrtf(ss);
    }
  }
}

__global__ void gate_finalize_kernel(const float* __restrict__ logits,
                                     const float* __restrict__ log_temp,
                                     const float* __restrict__ floor_param,
                                     float* __restrict__ p) {
  int idx = blockIdx.x * blockDim.x + threadIdx.x;
  if (idx >= TOKn * Hn) return;
  int h = idx & 3, tok = idx >> 2;
  const float* lg = logits + (size_t)tok * 16 + h * 4;
  float tinv = 1.f / expf(log_temp[h]);
  float x[4], mx = -3.0e38f;
#pragma unroll
  for (int s = 0; s < 4; ++s) { x[s] = lg[s] * tinv; mx = fmaxf(mx, x[s]); }
  float sum = 0.f;
#pragma unroll
  for (int s = 0; s < 4; ++s) { x[s] = expf(x[s] - mx); sum += x[s]; }
  float inv = 1.f / sum, cs = 0.f, cl[4];
#pragma unroll
  for (int s = 0; s < 4; ++s) {
    float fl = 0.075f * sigmoidf_(floor_param[h * 4 + s]);
    cl[s] = fmaxf(x[s] * inv, fl);
    cs += cl[s];
  }
  float* o = p + (size_t)tok * 16 + h * 4;
  float ci = 1.f / cs;
#pragma unroll
  for (int s = 0; s < 4; ++s) o[s] = cl[s] * ci;
}

__global__ void fuse_norm_kernel(const float* __restrict__ p, const float* __restrict__ sh,
                                 const float* __restrict__ lo, const float* __restrict__ de,
                                 const float* __restrict__ vd, const float* __restrict__ norm_w,
                                 float* __restrict__ o_fused) {
  int wid  = (blockIdx.x * blockDim.x + threadIdx.x) >> 5;
  int lane = threadIdx.x & 31;
  if (wid >= TOKn * Hn) return;
  int h = wid & 3, tok = wid >> 2;
  const float* pp = p + (size_t)tok * 16 + h * 4;
  float p0 = pp[0], p1 = pp[1], p2 = pp[2], p3 = pp[3];
  size_t base = (size_t)tok * Dn + h * DVn;
  float ov[8], ss = 0.f;
#pragma unroll
  for (int j = 0; j < 8; ++j) {
    int d = lane + 32 * j;
    float x = p0 * sh[base + d] + p1 * lo[base + d] + p2 * de[base + d] + p3 * vd[base + d];
    ov[j] = x; ss += x * x;
  }
#pragma unroll
  for (int o = 16; o; o >>= 1) ss += __shfl_xor(ss, o, 32);
  float inv = rsqrtf(ss * (1.f / 256.f) + 1e-5f);
#pragma unroll
  for (int j = 0; j < 8; ++j) {
    int d = lane + 32 * j;
    o_fused[base + d] = ov[j] * inv * norm_w[d];
  }
}

// ---------------------------------------------------------------- launch
extern "C" void kernel_launch(void* const* d_in, const int* in_sizes, int n_in,
                              void* d_out, int out_size, void* d_ws, size_t ws_size,
                              hipStream_t stream) {
  (void)in_sizes; (void)n_in; (void)out_size; (void)ws_size;
  const float* hidden   = (const float*)d_in[0];
  const float* Wq       = (const float*)d_in[1];
  const float* Wk       = (const float*)d_in[2];
  const float* Wv       = (const float*)d_in[3];
  const float* Wb       = (const float*)d_in[4];
  const float* forget_p = (const float*)d_in[5];
  const float* conv_q   = (const float*)d_in[6];
  const float* conv_k   = (const float*)d_in[7];
  const float* conv_v   = (const float*)d_in[8];
  const float* fir_s    = (const float*)d_in[9];
  const float* fir_l    = (const float*)d_in[10];
  const float* log_temp = (const float*)d_in[11];
  const float* floor_p  = (const float*)d_in[12];
  const float* W1       = (const float*)d_in[13];
  const float* b1       = (const float*)d_in[14];
  const float* W2       = (const float*)d_in[15];
  const float* b2       = (const float*)d_in[16];
  const float* norm_w   = (const float*)d_in[17];
  const float* Wo       = (const float*)d_in[18];
  float* out            = (float*)d_out;

  char* ws = (char*)d_ws;
  constexpr size_t MB = 1ull << 20;
  float*  qpre   = (float*)(ws + 0 * MB);     // reused later by gate_in/hmid/...
  float*  kpre   = (float*)(ws + 16 * MB);
  float*  vpre   = (float*)(ws + 32 * MB);
  float*  qact   = (float*)(ws + 48 * MB);
  float*  kact   = (float*)(ws + 64 * MB);
  float*  vact   = (float*)(ws + 80 * MB);    // v_direct (kept)
  float*  beta   = (float*)(ws + 96 * MB);
  __bf16* qn     = (__bf16*)(ws + 97 * MB);
  __bf16* kn     = (__bf16*)(ws + 105 * MB);
  __bf16* kbb    = (__bf16*)(ws + 113 * MB);
  __bf16* vbb    = (__bf16*)(ws + 121 * MB);
  __bf16* u_g    = (__bf16*)(ws + 129 * MB);
  __bf16* w_g    = (__bf16*)(ws + 137 * MB);
  float*  delta  = (float*)(ws + 145 * MB);
  float*  shortb = (float*)(ws + 161 * MB);
  float*  longb  = (float*)(ws + 177 * MB);   // high-water 193 MB
  // region reuse (q/k pre+post are dead by the time these are written):
  float*  gatein = (float*)(ws + 0 * MB);     // 4096x1088
  float*  hmid   = (float*)(ws + 18 * MB);    // 4096x2048
  float*  logits = (float*)(ws + 52 * MB);    // 4096x16
  float*  pgate  = (float*)(ws + 53 * MB);
  float*  ofused = (float*)(ws + 54 * MB);    // 4096x1024

  dim3 blk(256);
  auto gemm_grid = [](int M, int N) { return dim3((unsigned)((N + 63) / 64), (unsigned)((M + 63) / 64)); };

  // 1) q/k/v projections
  gemm_bf16_kernel<<<gemm_grid(TOKn, Dn), blk, 0, stream>>>(hidden, Wq, nullptr, qpre, TOKn, Dn, Dn, 0);
  gemm_bf16_kernel<<<gemm_grid(TOKn, Dn), blk, 0, stream>>>(hidden, Wk, nullptr, kpre, TOKn, Dn, Dn, 0);
  gemm_bf16_kernel<<<gemm_grid(TOKn, Dn), blk, 0, stream>>>(hidden, Wv, nullptr, vpre, TOKn, Dn, Dn, 0);

  // 2) causal dwconv(4) + silu
  int ew = (TOKn * Dn) / 256;
  dwconv4_silu_kernel<<<ew, blk, 0, stream>>>(qpre, conv_q, qact);
  dwconv4_silu_kernel<<<ew, blk, 0, stream>>>(kpre, conv_k, kact);
  dwconv4_silu_kernel<<<ew, blk, 0, stream>>>(vpre, conv_v, vact);

  // 3) beta, 4) l2norm + head split (bf16)
  beta_kernel<<<(TOKn * Hn * 32) / 256, blk, 0, stream>>>(hidden, Wb, beta);
  qknorm_kernel<<<(TOKn * Hn * 32) / 256, blk, 0, stream>>>(qact, kact, vact, beta, qn, kn, kbb, vbb);

  // 5) per-chunk T/u/w, 6) stateful scan
  chunk_prepare_kernel<<<Bn * Hn * NCHn, blk, 0, stream>>>(kn, kbb, vbb, u_g, w_g);
  delta_scan_kernel<<<Bn * Hn * 8, blk, 0, stream>>>(qn, kn, u_g, w_g, forget_p, delta);

  // 7) FIR branches on v_direct
  fir_kernel<<<ew, blk, 0, stream>>>(vact, fir_s, shortb, 3);
  fir_kernel<<<ew, blk, 0, stream>>>(vact, fir_l, longb, 63);

  // 8) gate input (hidden ++ per-head stats), 9) MLP
  gate_build_kernel<<<TOKn, blk, 0, stream>>>(hidden, shortb, longb, delta, vact, gatein);
  gemm_bf16_kernel<<<gemm_grid(TOKn, 2 * Dn), blk, 0, stream>>>(gatein, W1, b1, hmid, TOKn, 2 * Dn, 1088, 1);
  gemm_bf16_kernel<<<gemm_grid(TOKn, 16), blk, 0, stream>>>(hmid, W2, b2, logits, TOKn, 16, 2 * Dn, 0);
  gate_finalize_kernel<<<(TOKn * Hn) / 256, blk, 0, stream>>>(logits, log_temp, floor_p, pgate);

  // 10) fuse + rmsnorm, 11) output projection
  fuse_norm_kernel<<<(TOKn * Hn * 32) / 256, blk, 0, stream>>>(pgate, shortb, longb, delta, vact, norm_w, ofused);
  gemm_bf16_kernel<<<gemm_grid(TOKn, Dn), blk, 0, stream>>>(ofused, Wo, nullptr, out, TOKn, Dn, Dn, 0);
}
